// q_pi_class_5772436046288
// MI455X (gfx1250) — matrix operand order repaired and checked by reading, not compiled, not run on hardware
//
#include <hip/hip_runtime.h>
#include <hip/hip_bf16.h>

typedef _Float16 v16h __attribute__((ext_vector_type(16)));
typedef _Float16 v8h  __attribute__((ext_vector_type(8)));
typedef float    v8f  __attribute__((ext_vector_type(8)));

#define NROW 524288
#define KCB  512
#define DIM  64
#define ROWS_PER_BLOCK 128
#define NBLOCKS (NROW / ROWS_PER_BLOCK)   // 4096
#define NPARTIALS (NBLOCKS * 8)           // one per wave

#if __has_builtin(__builtin_amdgcn_global_load_async_to_lds_b128) && \
    __has_builtin(__builtin_amdgcn_s_wait_asynccnt)
#define HAS_ASYNC_LDS 1
typedef int v4i __attribute__((vector_size(16)));            // matches builtin param
typedef __attribute__((address_space(1))) v4i gv4i_t;        // global int4
typedef __attribute__((address_space(3))) v4i lv4i_t;        // LDS int4
#else
#define HAS_ASYNC_LDS 0
#endif

// ---------------------------------------------------------------------------
// Kernel A: precompute ||e||^2 and the f16, K-major, XOR-swizzled codebook.
// Logical layout: EB[k][c]; the 16-byte chunk index j (= c/8) is remapped to
// j ^ (k & 15) so per-lane B-fragment 16-bit LDS reads are bank-conflict-free.
// ---------------------------------------------------------------------------
__global__ __launch_bounds__(512) void vq_prep(const float* __restrict__ cb,
                                               _Float16* __restrict__ eb,
                                               float* __restrict__ norms) {
    int c = threadIdx.x;              // one code per thread, 512 threads
    const float* row = cb + c * DIM;
    float s = 0.f;
#pragma unroll
    for (int k = 0; k < DIM; ++k) { float v = row[k]; s += v * v; }
    norms[c] = s;

    int cj = c >> 3;                  // 16B chunk within a K-row
    int co = c & 7;                   // f16 slot within chunk
#pragma unroll
    for (int k = 0; k < DIM; ++k) {
        int jp = cj ^ (k & 15);
        eb[k * KCB + jp * 8 + co] = (_Float16)row[k];
    }
}

// ---------------------------------------------------------------------------
// Kernel B: per wave: 16 rows of X. Stage codebook (64 KB) in LDS (async if
// available), distance GEMM via v_wmma_f32_16x16x32_f16, argmin over 512
// codes, exact fp32 gather + fused loss partial.
// ---------------------------------------------------------------------------
__global__ __launch_bounds__(256) void vq_main(const float* __restrict__ x,
                                               const float* __restrict__ cb,
                                               const _Float16* __restrict__ eb_ws,
                                               const float* __restrict__ norms,
                                               float* __restrict__ out_idx,
                                               float* __restrict__ out_q,
                                               float* __restrict__ partials) {
    __shared__ _Float16 eb[DIM * KCB];              // exactly 65536 bytes

    const int lane   = threadIdx.x & 31;
    const int wave   = threadIdx.x >> 5;
    const int half   = lane >> 4;
    const int lanelo = lane & 15;
    const size_t m0  = (size_t)blockIdx.x * ROWS_PER_BLOCK + (size_t)wave * 16;

    const float* xrow = x + (m0 + (size_t)lanelo) * DIM;
    __builtin_prefetch(xrow, 0, 0);                 // gfx1250 global_prefetch_b8

    // cooperative 64KB codebook staging, 16 x b128 per thread
#if HAS_ASYNC_LDS
    {
        gv4i_t* gsrc = (gv4i_t*)eb_ws;              // generic -> AS(1) int4*
        lv4i_t* ldst = (lv4i_t*)eb;                 // generic -> AS(3) int4*
        for (int i = threadIdx.x; i < (DIM * KCB) / 8; i += 256) {
            __builtin_amdgcn_global_load_async_to_lds_b128(gsrc + i, ldst + i, 0, 0);
        }
        __builtin_amdgcn_s_wait_asynccnt(0);
    }
#else
    {
        const uint4* src = (const uint4*)eb_ws;
        uint4* dst = (uint4*)eb;
        for (int i = threadIdx.x; i < (DIM * KCB) / 8; i += 256) dst[i] = src[i];
    }
#endif
    __syncthreads();

    // --- A fragments (16x32 f16, ISA per-lane layout) ------------------------
    v16h a0, a1;
#pragma unroll
    for (int v = 0; v < 8; ++v) {
        int kk = ((v < 4) ? 0 : 16) + half * 8 + (v & 3) * 2;
        float2 f0 = *(const float2*)(xrow + kk);
        float2 f1 = *(const float2*)(xrow + 32 + kk);
        a0[2 * v]     = (_Float16)f0.x;  a0[2 * v + 1] = (_Float16)f0.y;
        a1[2 * v]     = (_Float16)f1.x;  a1[2 * v + 1] = (_Float16)f1.y;
    }

    float minv[8];
    int   mini[8];
#pragma unroll
    for (int r = 0; r < 8; ++r) { minv[r] = 3.4e38f; mini[r] = 0; }

    const int k0 = lane;        // B chunk 0: this lane's K-row
    const int k1 = 32 + lane;   // B chunk 1
    const v8h* ebv = (const v8h*)eb;

#pragma unroll 2
    for (int c = 0; c < KCB / 16; ++c) {            // 32 blocks of 16 codes
        int nb = c * 16 + lanelo;
        int nj = nb >> 3, no = nb & 7;
        v16h b0, b1;
#pragma unroll
        for (int e = 0; e < 8; ++e) {
            int ka = half * 8 + e, kb = 16 + half * 8 + e;
            b0[e]     = eb[ka * KCB + ((nj ^ (ka & 15)) << 3) + no];
            b0[e + 8] = eb[kb * KCB + ((nj ^ (kb & 15)) << 3) + no];
            b1[e]     = eb[(ka + 32) * KCB + ((nj ^ ((ka + 32) & 15)) << 3) + no];
            b1[e + 8] = eb[(kb + 32) * KCB + ((nj ^ ((kb + 32) & 15)) << 3) + no];
        }
        v8f acc = {};
        acc = __builtin_amdgcn_wmma_f32_16x16x32_f16(false, a0, false, b0,
                                                     (short)0, acc, false, false);
        acc = __builtin_amdgcn_wmma_f32_16x16x32_f16(false, a1, false, b1,
                                                     (short)0, acc, false, false);
        float en  = norms[c * 16 + lanelo];
        int  code = c * 16 + lanelo;
#pragma unroll
        for (int r = 0; r < 8; ++r) {
            float dist = en - 2.0f * acc[r];        // + ||x||^2 is argmin-invariant
            if (dist < minv[r]) { minv[r] = dist; mini[r] = code; }
        }
    }

    // reduce argmin across the 16 lanes of each C-column group
#pragma unroll
    for (int mask = 1; mask <= 8; mask <<= 1) {
#pragma unroll
        for (int r = 0; r < 8; ++r) {
            float ov = __shfl_xor(minv[r], mask, 32);
            int   oi = __shfl_xor(mini[r], mask, 32);
            if (ov < minv[r] || (ov == minv[r] && oi < mini[r])) {
                minv[r] = ov; mini[r] = oi;
            }
        }
    }
    // lane 0 now holds rows m0..m0+7 (r=0..7), lane 16 holds rows m0+8..m0+15

    // --- gather + store + loss partial (exact fp32) --------------------------
    float lsum = 0.f;
#pragma unroll
    for (int m = 0; m < 16; ++m) {
        int idxm = __shfl(mini[m & 7], (m >> 3) * 16, 32);
        size_t row = m0 + (size_t)m;
        if (lane == 0) out_idx[row] = (float)idxm;
        float2 q2 = *(const float2*)(cb + (size_t)idxm * DIM + lane * 2);
        float2 x2 = *(const float2*)(x + row * DIM + lane * 2);
        float dx = q2.x - x2.x, dy = q2.y - x2.y;
        lsum += dx * dx + dy * dy;
        *(float2*)(out_q + row * DIM + lane * 2) = q2;   // quantized_st == quantized
    }
#pragma unroll
    for (int mask = 1; mask < 32; mask <<= 1) lsum += __shfl_xor(lsum, mask, 32);
    if (lane == 0) partials[blockIdx.x * 8 + wave] = lsum;
}

// ---------------------------------------------------------------------------
// Kernel C: deterministic tree reduce of the per-wave loss partials.
// vq_loss = 0.25*mean + mean = 1.25 * sum((q-x)^2) / (N*D)
// ---------------------------------------------------------------------------
__global__ __launch_bounds__(256) void vq_loss_reduce(const float* __restrict__ partials,
                                                      int n, float* __restrict__ out_loss) {
    __shared__ float s[256];
    float t = 0.f;
    for (int i = threadIdx.x; i < n; i += 256) t += partials[i];
    s[threadIdx.x] = t;
    __syncthreads();
    for (int st = 128; st > 0; st >>= 1) {
        if ((int)threadIdx.x < st) s[threadIdx.x] += s[threadIdx.x + st];
        __syncthreads();
    }
    if (threadIdx.x == 0)
        out_loss[0] = 1.25f * s[0] / (float)((double)NROW * (double)DIM);
}

extern "C" void kernel_launch(void* const* d_in, const int* in_sizes, int n_in,
                              void* d_out, int out_size, void* d_ws, size_t ws_size,
                              hipStream_t stream) {
    const float* x  = (const float*)d_in[0];   // [N, D] f32
    const float* cb = (const float*)d_in[1];   // [K, D] f32

    float* out      = (float*)d_out;
    float* out_idx  = out;                       // [N]    (indices as float)
    float* out_q    = out + NROW;                // [N, D]
    float* out_loss = out + NROW + (size_t)NROW * DIM;  // [1]

    char* ws = (char*)d_ws;
    _Float16* eb_ws  = (_Float16*)ws;                        // 65536 B
    float*    norms  = (float*)(ws + 65536);                 // 2048 B
    float*    parts  = (float*)(ws + 65536 + 2048);          // 131072 B

    vq_prep<<<1, 512, 0, stream>>>(cb, eb_ws, norms);
    vq_main<<<NBLOCKS, 256, 0, stream>>>(x, cb, eb_ws, norms, out_idx, out_q, parts);
    vq_loss_reduce<<<1, 256, 0, stream>>>(parts, NPARTIALS, out_loss);
}
